// ProgressiveWindowAttention_90082644066320
// MI455X (gfx1250) — compile-verified, matrix operand order checked by reading, not run on hardware
//
#include <hip/hip_runtime.h>

typedef __attribute__((ext_vector_type(16))) _Float16 v16h;
typedef __attribute__((ext_vector_type(8)))  float    v8f;

// Problem dims (compile-time)
constexpr int  Bn   = 2048;           // windows
constexpr int  Hn   = 16;             // heads
constexpr int  Cn   = 512;            // channels
constexpr int  C3n  = 3 * Cn;         // 1536
constexpr int  Ntok = 64;             // tokens per window
constexpr float SCALE = 0.17677669529663689f; // 1/sqrt(32)

// ---------------------------------------------------------------------------
// Async global->LDS copy (CDNA5 GLOBAL_LOAD_ASYNC_TO_LDS_B128, ASYNCcnt).
// Builtin signature (from hipcc diagnostic): first param is
// 'int __attribute__((vector_size(16))) __device__ *' (global int4 vec ptr).
// ---------------------------------------------------------------------------
#if defined(__gfx1250__) &&                                            \
    __has_builtin(__builtin_amdgcn_global_load_async_to_lds_b128) &&   \
    __has_builtin(__builtin_amdgcn_s_wait_asynccnt)
#define HAVE_ASYNC_LDS 1
#else
#define HAVE_ASYNC_LDS 0
#endif

#if HAVE_ASYNC_LDS
typedef int vec4i __attribute__((vector_size(16)));
typedef __attribute__((address_space(1))) vec4i* glb_v4i_p;
typedef __attribute__((address_space(3))) vec4i* lds_v4i_p;
#endif

__device__ __forceinline__ void async_copy_b128(const _Float16* g, _Float16* l) {
#if HAVE_ASYNC_LDS
  __builtin_amdgcn_global_load_async_to_lds_b128((glb_v4i_p)g, (lds_v4i_p)l, 0, 0);
#else
  *(uint4*)l = *(const uint4*)g;
#endif
}

__device__ __forceinline__ void async_wait_all() {
#if HAVE_ASYNC_LDS
  __builtin_amdgcn_s_wait_asynccnt(0);
#endif
}

// ---------------------------------------------------------------------------
// f32 -> f16 cast, 4 elements / thread (all sizes are multiples of 4)
// ---------------------------------------------------------------------------
__global__ void __launch_bounds__(256) cvt_f32_to_f16(const float* __restrict__ src,
                                                      _Float16* __restrict__ dst, long n4) {
  long i = (long)blockIdx.x * blockDim.x + threadIdx.x;
  if (i >= n4) return;
  const float4 f = *(const float4*)(src + i * 4);
  union { _Float16 h[4]; uint2 u; } o;
  o.h[0] = (_Float16)f.x; o.h[1] = (_Float16)f.y;
  o.h[2] = (_Float16)f.z; o.h[3] = (_Float16)f.w;
  *(uint2*)(dst + i * 4) = o.u;
}

// ---------------------------------------------------------------------------
// Tiled f16 WMMA GEMM: C[M,Nc] = A[M,K] * B[K,Nc] + bias[Nc]
// BM=128 BN=128 BK=64, 256 threads = 8 waves, each wave owns 32x64 (2x4 tiles).
// Double-buffered LDS; A staged with async b128 copies, B staged transposed
// via a 4(k)x8(n) register transpose -> ds_store_b64.
// ---------------------------------------------------------------------------
template <bool OUT_F16>
__global__ void __launch_bounds__(256) gemm_f16_wmma(const _Float16* __restrict__ A,
                                                     const _Float16* __restrict__ Bm,
                                                     const float* __restrict__ bias,
                                                     void* __restrict__ Cout,
                                                     int M, int Ncols, int K) {
  __shared__ __align__(32) _Float16 As[2][128][64];   // [m][k]
  __shared__ __align__(32) _Float16 Bs[2][128][64];   // [n][k] (transposed stage)

  const int t    = threadIdx.x;
  const int lane = t & 31;
  const int wid  = t >> 5;         // 0..7
  const int wm   = wid & 3;        // M strip (32 rows each)
  const int wn   = wid >> 2;       // N strip (64 cols each)
  const int m0   = blockIdx.x * 128;
  const int n0   = blockIdx.y * 128;

  auto stageA = [&](int k0, int buf) {
    // 128 rows x 64 halves = 1024 b128 copies, 4 per thread
    for (int q = t; q < 1024; q += 256) {
      const int row = q >> 3, seg = q & 7;
      async_copy_b128(A + (long)(m0 + row) * K + k0 + seg * 8, &As[buf][row][seg * 8]);
    }
  };
  auto stageB = [&](int k0, int buf) {
    // one 4(k) x 8(n) patch per thread: 4x b128 loads -> 8x ds_store_b64
    const int pk = t & 15;         // k block of 4  (covers 64)
    const int pn = t >> 4;         // n block of 8  (covers 128)
    union { uint4 u; ushort s[8]; } r[4];
    for (int kk = 0; kk < 4; ++kk)
      r[kk].u = *(const uint4*)(Bm + (long)(k0 + pk * 4 + kk) * Ncols + n0 + pn * 8);
    for (int j = 0; j < 8; ++j) {
      union { ushort s[4]; uint2 u; } w;
      w.s[0] = r[0].s[j]; w.s[1] = r[1].s[j]; w.s[2] = r[2].s[j]; w.s[3] = r[3].s[j];
      *(uint2*)&Bs[buf][pn * 8 + j][pk * 4] = w.u;
    }
  };

  v8f acc[2][4];
  for (int i = 0; i < 2; ++i)
    for (int j = 0; j < 4; ++j) acc[i][j] = {};

  stageA(0, 0);
  stageB(0, 0);
  async_wait_all();
  __syncthreads();

  const int ksteps = K / 64;
  int buf = 0;
  for (int s = 0; s < ksteps; ++s) {
    if (s + 1 < ksteps) {           // overlap next-tile staging with compute
      stageA((s + 1) * 64, buf ^ 1);
      stageB((s + 1) * 64, buf ^ 1);
    }
    for (int kk = 0; kk < 2; ++kk) {
      v16h af[2], bf[4];
      for (int i = 0; i < 2; ++i)
        af[i] = *(const v16h*)&As[buf][wm * 32 + i * 16 + (lane & 15)]
                                      [kk * 32 + (lane >> 4) * 16];
      for (int j = 0; j < 4; ++j)
        bf[j] = *(const v16h*)&Bs[buf][wn * 64 + j * 16 + (lane & 15)]
                                      [kk * 32 + (lane >> 4) * 16];
      for (int i = 0; i < 2; ++i)
        for (int j = 0; j < 4; ++j)
          acc[i][j] = __builtin_amdgcn_wmma_f32_16x16x32_f16(
              false, af[i], false, bf[j], (short)0, acc[i][j], false, false);
    }
    async_wait_all();
    __syncthreads();
    buf ^= 1;
  }

  // C/D layout: VGPR r, lanes 0-15 -> M=r, lanes 16-31 -> M=r+8; N = lane&15
  for (int i = 0; i < 2; ++i) {
    for (int j = 0; j < 4; ++j) {
      const int col = n0 + wn * 64 + j * 16 + (lane & 15);
      const float bv = bias[col];
      for (int r = 0; r < 8; ++r) {
        const int row = m0 + wm * 32 + i * 16 + r + (lane >> 4) * 8;
        const float val = acc[i][j][r] + bv;
        if constexpr (OUT_F16)
          ((_Float16*)Cout)[(long)row * Ncols + col] = (_Float16)val;
        else
          ((float*)Cout)[(long)row * Ncols + col] = val;
      }
    }
  }
}

// ---------------------------------------------------------------------------
// Fused window attention: one 4-wave workgroup per (b, h).
// S = (Q Kt) * SCALE + rpb[relidx] + mask[b%64]; P = softmax(S); O = P V
// ---------------------------------------------------------------------------
__global__ void __launch_bounds__(128) attn_wmma(const _Float16* __restrict__ qkvh,
                                                 const float* __restrict__ mask,
                                                 const float* __restrict__ rpb,
                                                 _Float16* __restrict__ aout) {
  __shared__ __align__(32) _Float16 Qs[64][32];
  __shared__ __align__(32) _Float16 Ks[64][32];
  __shared__ __align__(32) _Float16 Vt[32][64];   // V transposed: [hd][token]
  __shared__ __align__(32) float    Sm[64][64];
  __shared__ __align__(32) _Float16 Pm[64][64];

  const int t    = threadIdx.x;
  const int lane = t & 31;
  const int wid  = t >> 5;                 // 0..3, owns 16-row strip
  const int b    = blockIdx.x >> 4;
  const int h    = blockIdx.x & 15;
  const long base = ((long)b * Ntok) * C3n + (long)h * 32;

  // Stage Q, K via async b128 (layout-preserving), 2 rows' worth per thread.
  for (int q = t; q < 256; q += 128) {
    const int row = q >> 2, seg = q & 3;
    const long rb = base + (long)row * C3n + seg * 8;
    async_copy_b128(qkvh + rb,      &Qs[row][seg * 8]);     // s=0
    async_copy_b128(qkvh + rb + Cn, &Ks[row][seg * 8]);     // s=1
  }
  // Stage V transposed: 64 patches of 4(token) x 8(hd) on threads t<64.
  if (t < 64) {
    const int ptk = t & 15;       // token block of 4 (covers 64)
    const int phd = t >> 4;       // hd block of 8    (covers 32)
    union { uint4 u; ushort s[8]; } r[4];
    for (int kk = 0; kk < 4; ++kk)
      r[kk].u = *(const uint4*)(qkvh + base + (long)(ptk * 4 + kk) * C3n + 2 * Cn + phd * 8);
    for (int j = 0; j < 8; ++j) {
      union { ushort s[4]; uint2 u; } w;
      w.s[0] = r[0].s[j]; w.s[1] = r[1].s[j]; w.s[2] = r[2].s[j]; w.s[3] = r[3].s[j];
      *(uint2*)&Vt[phd * 8 + j][ptk * 4] = w.u;
    }
  }
  async_wait_all();
  __syncthreads();

  const int m0 = wid * 16;

  // ---- S = Q Kt (4 WMMAs per wave, K=32 in one shot) ----
  {
    const v16h aq = *(const v16h*)&Qs[m0 + (lane & 15)][(lane >> 4) * 16];
    v8f sa[4]; for (int j = 0; j < 4; ++j) sa[j] = {};
    for (int j = 0; j < 4; ++j) {
      const v16h bk = *(const v16h*)&Ks[j * 16 + (lane & 15)][(lane >> 4) * 16];
      sa[j] = __builtin_amdgcn_wmma_f32_16x16x32_f16(
          false, aq, false, bk, (short)0, sa[j], false, false);
    }
    for (int j = 0; j < 4; ++j)
      for (int r = 0; r < 8; ++r)
        Sm[m0 + r + (lane >> 4) * 8][j * 16 + (lane & 15)] = sa[j][r] * SCALE;
  }
  __syncthreads();

  // ---- bias + mask + softmax: lane owns row m0+(lane&15), half (lane>>4) ----
  {
    const int row = m0 + (lane & 15);
    const int ch  = (lane >> 4) * 32;
    const int i1  = row >> 3, j1 = row & 7;      // W=8 coords of query token
    const float* mrow = mask + ((long)(b & 63) * Ntok + row) * Ntok;
    float vals[32];
    float mx = -3.0e38f;
    for (int c = 0; c < 32; ++c) {
      const int col = ch + c;
      const int i2 = col >> 3, j2 = col & 7;
      const int ridx = (i1 - i2 + 7) * 15 + (j1 - j2 + 7);
      const float v = Sm[row][col] + rpb[ridx * Hn + h] + mrow[col];
      vals[c] = v;
      mx = fmaxf(mx, v);
    }
    mx = fmaxf(mx, __shfl_xor(mx, 16, 32));      // join the two half-rows
    float sum = 0.f;
    for (int c = 0; c < 32; ++c) { vals[c] = __expf(vals[c] - mx); sum += vals[c]; }
    sum += __shfl_xor(sum, 16, 32);
    const float inv = 1.0f / sum;
    for (int c = 0; c < 32; ++c) Pm[row][ch + c] = (_Float16)(vals[c] * inv);
  }
  __syncthreads();

  // ---- O = P V (16x32 per wave: 2 N-tiles x 2 K-chunks) ----
  {
    v8f oa[2]; oa[0] = {}; oa[1] = {};
    for (int kk = 0; kk < 2; ++kk) {
      const v16h ap = *(const v16h*)&Pm[m0 + (lane & 15)][kk * 32 + (lane >> 4) * 16];
      for (int j = 0; j < 2; ++j) {
        const v16h bv = *(const v16h*)&Vt[j * 16 + (lane & 15)][kk * 32 + (lane >> 4) * 16];
        oa[j] = __builtin_amdgcn_wmma_f32_16x16x32_f16(
            false, ap, false, bv, (short)0, oa[j], false, false);
      }
    }
    // write bnhd layout: aout[b, n, h*32 + hd] in f16 (feeds proj GEMM)
    for (int j = 0; j < 2; ++j)
      for (int r = 0; r < 8; ++r) {
        const int row = m0 + r + (lane >> 4) * 8;
        const int hd  = j * 16 + (lane & 15);
        aout[((long)b * Ntok + row) * Cn + h * 32 + hd] = (_Float16)oa[j][r];
      }
  }
}

// ---------------------------------------------------------------------------
extern "C" void kernel_launch(void* const* d_in, const int* in_sizes, int n_in,
                              void* d_out, int out_size, void* d_ws, size_t ws_size,
                              hipStream_t stream) {
  (void)in_sizes; (void)n_in; (void)out_size; (void)ws_size;
  const float* x      = (const float*)d_in[0];
  const float* mask   = (const float*)d_in[1];
  const float* qkv_w  = (const float*)d_in[2];
  const float* qkv_b  = (const float*)d_in[3];
  const float* rpb    = (const float*)d_in[4];
  const float* proj_w = (const float*)d_in[5];
  const float* proj_b = (const float*)d_in[6];
  float* out = (float*)d_out;

  const long MX = (long)Bn * Ntok;          // 131072 rows
  char* ws = (char*)d_ws;
  _Float16* xh   = (_Float16*)ws; ws += MX * Cn * sizeof(_Float16);
  _Float16* qwh  = (_Float16*)ws; ws += (long)Cn * C3n * sizeof(_Float16);
  _Float16* pwh  = (_Float16*)ws; ws += (long)Cn * Cn * sizeof(_Float16);
  _Float16* qkvh = (_Float16*)ws; ws += MX * C3n * sizeof(_Float16);
  _Float16* aoh  = (_Float16*)ws; ws += MX * Cn * sizeof(_Float16);

  // 1) downcast inputs to f16
  {
    const long n4x = MX * Cn / 4;
    cvt_f32_to_f16<<<(unsigned)((n4x + 255) / 256), 256, 0, stream>>>(x, xh, n4x);
    const long n4q = (long)Cn * C3n / 4;
    cvt_f32_to_f16<<<(unsigned)((n4q + 255) / 256), 256, 0, stream>>>(qkv_w, qwh, n4q);
    const long n4p = (long)Cn * Cn / 4;
    cvt_f32_to_f16<<<(unsigned)((n4p + 255) / 256), 256, 0, stream>>>(proj_w, pwh, n4p);
  }

  // 2) qkv = x @ qkv_w + qkv_b   -> f16 [131072, 1536]
  gemm_f16_wmma<true><<<dim3(MX / 128, C3n / 128), 256, 0, stream>>>(
      xh, qwh, qkv_b, (void*)qkvh, (int)MX, C3n, Cn);

  // 3) fused window attention -> f16 [131072, 512] in bnhd layout
  attn_wmma<<<Bn * Hn, 128, 0, stream>>>(qkvh, mask, rpb, aoh);

  // 4) out = attn_out @ proj_w + proj_b -> f32 d_out
  gemm_f16_wmma<false><<<dim3(MX / 128, Cn / 128), 256, 0, stream>>>(
      aoh, pwh, proj_b, (void*)out, (int)MX, Cn, Cn);
}
